// LSTM_69114613728647
// MI455X (gfx1250) — matrix operand order repaired and checked by reading, not compiled, not run on hardware
//
#include <hip/hip_runtime.h>
#include <math.h>

typedef __attribute__((ext_vector_type(16))) __bf16 v16bf;
typedef __attribute__((ext_vector_type(8)))  float  v8f;
typedef __attribute__((ext_vector_type(4)))  unsigned int uint4v;
typedef unsigned short u16;

union FragBF {
    v16bf v;
    uint4v q[2];   // 32 bytes = 16 bf16
};

union PackBF4 {
    u16 h[4];
    unsigned long long u;
};

__device__ __forceinline__ u16 f2bf(float f) {
    unsigned int u = __float_as_uint(f);
    unsigned int r = (u + 0x7FFFu + ((u >> 16) & 1u)) >> 16;   // RNE
    return (u16)r;
}

__device__ __forceinline__ float sigmoidf_fast(float z) {
    return 1.0f / (1.0f + __expf(-z));
}

// ---------------------------------------------------------------------------
// Prep: convert Wh (4 gates, each 1024x1024) and Wph (128x1024) to bf16,
// build gate-major packed x-weights Wxp[4][1024] and fused bias
// bsum[4][1024] = bx_g + bh_g, zero h ping-pong buffers and c.
// ---------------------------------------------------------------------------
__global__ void prep_kernel(const float* __restrict__ Wfh, const float* __restrict__ Wih,
                            const float* __restrict__ Wgh, const float* __restrict__ Woh,
                            const float* __restrict__ Wph,
                            const float* __restrict__ Wfx, const float* __restrict__ Wix,
                            const float* __restrict__ Wgx, const float* __restrict__ Wox,
                            const float* __restrict__ bfx, const float* __restrict__ bix,
                            const float* __restrict__ bgx, const float* __restrict__ box_,
                            const float* __restrict__ bfh, const float* __restrict__ bih,
                            const float* __restrict__ bgh, const float* __restrict__ boh,
                            u16* __restrict__ Whb, u16* __restrict__ Wpb,
                            float* __restrict__ Wxp, float* __restrict__ bsum,
                            u16* __restrict__ hA, u16* __restrict__ hB,
                            float* __restrict__ c) {
    size_t i = (size_t)blockIdx.x * blockDim.x + threadIdx.x;
    size_t stride = (size_t)gridDim.x * blockDim.x;
    const size_t NW = (size_t)4 * 1024 * 1024;
    for (size_t idx = i; idx < NW; idx += stride) {
        int g = (int)(idx >> 20);
        size_t r = idx & 0xFFFFFu;
        float v = (g == 0) ? Wfh[r] : (g == 1) ? Wih[r] : (g == 2) ? Wgh[r] : Woh[r];
        Whb[idx] = f2bf(v);
    }
    for (size_t idx = i; idx < (size_t)128 * 1024; idx += stride)
        Wpb[idx] = f2bf(Wph[idx]);
    for (size_t idx = i; idx < (size_t)4 * 1024; idx += stride) {
        int g = (int)(idx >> 10);
        size_t j = idx & 1023u;
        float wx = (g == 0) ? Wfx[j] : (g == 1) ? Wix[j] : (g == 2) ? Wgx[j] : Wox[j];
        float bx = (g == 0) ? bfx[j] : (g == 1) ? bix[j] : (g == 2) ? bgx[j] : box_[j];
        float bh = (g == 0) ? bfh[j] : (g == 1) ? bih[j] : (g == 2) ? bgh[j] : boh[j];
        Wxp[idx]  = wx;
        bsum[idx] = bx + bh;
    }
    for (size_t idx = i; idx < (size_t)256 * 1024; idx += stride) {
        hA[idx] = 0; hB[idx] = 0; c[idx] = 0.0f;
    }
}

// ---------------------------------------------------------------------------
// One LSTM timestep.
// Block = 128 threads = 4 waves; wave g computes gate g's 16x32 GEMM tile.
// Grid = (256/16) * (1024/32) = 512 blocks.
// z_g = h_prev @ Wh_g^T (WMMA bf16, fp32 acc), then in LDS combine:
// z += x[t,b]*Wxp[g][j] + bsum[g][j]; LSTM cell update (vectorized x4).
// ---------------------------------------------------------------------------
__global__ __launch_bounds__(128)
void lstm_step(int t, const float* __restrict__ x,
               const u16* __restrict__ hin, u16* __restrict__ hout,
               float* __restrict__ c, const u16* __restrict__ Whb,
               const float* __restrict__ Wxp, const float* __restrict__ bsum) {
    __shared__ float zs[4][16 * 32];

    const int lane = threadIdx.x & 31;
    const int g    = threadIdx.x >> 5;          // gate index: 0=f,1=i,2=g,3=o
    const int mt   = blockIdx.x & 15;           // batch tile
    const int nt   = blockIdx.x >> 4;           // hidden tile
    const int m_base = mt * 16;
    const int n_base = nt * 32;

    v8f acc0 = {};
    v8f acc1 = {};

    // A fragment (16x32 bf16): lanes 0-15 row M=lane, K {0..15};
    // lanes 16-31 row M=lane-16, K {16..31}.
    const u16* aptr = hin + (size_t)(m_base + (lane & 15)) * 1024 + ((lane >> 4) << 4);
    // B fragment (32x16 bf16): lane j -> column N=j&15, K half = (lane>>4)*16.
    // Wh row-major (col n of B = row n of Wh) -> contiguous K run.
    const u16* bptr0 = Whb + ((size_t)g << 20)
                     + (size_t)(n_base + (lane & 15)) * 1024 + ((lane >> 4) << 4);
    const u16* bptr1 = bptr0 + (size_t)16 * 1024;

    #pragma unroll 4
    for (int k0 = 0; k0 < 1024; k0 += 32) {
        FragBF a, b0, b1;
        a.q[0]  = *(const uint4v*)(aptr  + k0);
        a.q[1]  = *(const uint4v*)(aptr  + k0 + 8);
        b0.q[0] = *(const uint4v*)(bptr0 + k0);
        b0.q[1] = *(const uint4v*)(bptr0 + k0 + 8);
        b1.q[0] = *(const uint4v*)(bptr1 + k0);
        b1.q[1] = *(const uint4v*)(bptr1 + k0 + 8);
        acc0 = __builtin_amdgcn_wmma_f32_16x16x32_bf16(false, a.v, false, b0.v,
                                                       (short)0, acc0, false, false);
        acc1 = __builtin_amdgcn_wmma_f32_16x16x32_bf16(false, a.v, false, b1.v,
                                                       (short)0, acc1, false, false);
    }

    // C/D layout: VGPR v, lanes0-15 -> M=v, lanes16-31 -> M=v+8; N = lane&15.
    {
        const int mh = (lane >> 4) << 3;
        const int nl = lane & 15;
        #pragma unroll
        for (int v = 0; v < 8; ++v) {
            zs[g][(v + mh) * 32 + nl]      = acc0[v];
            zs[g][(v + mh) * 32 + nl + 16] = acc1[v];
        }
    }
    __syncthreads();

    // Elementwise cell update: each thread owns 4 consecutive hidden units
    // so c traffic is b128 and the h store is one b64 of packed bf16.
    {
        const int ml  = threadIdx.x >> 3;          // 0..15  (batch row in tile)
        const int nl0 = (threadIdx.x & 7) << 2;    // 0,4,...,28
        const int b   = m_base + ml;
        const int j   = n_base + nl0;
        const float xv = x[(size_t)t * 256 + b];

        float zv[4][4];
        #pragma unroll
        for (int gg = 0; gg < 4; ++gg) {
            const float4 wv = *(const float4*)(Wxp  + (gg << 10) + j);
            const float4 bv = *(const float4*)(bsum + (gg << 10) + j);
            const float* zrow = &zs[gg][ml * 32 + nl0];
            zv[gg][0] = zrow[0] + xv * wv.x + bv.x;
            zv[gg][1] = zrow[1] + xv * wv.y + bv.y;
            zv[gg][2] = zrow[2] + xv * wv.z + bv.z;
            zv[gg][3] = zrow[3] + xv * wv.w + bv.w;
        }

        const size_t off = (size_t)b * 1024 + j;
        float4 cold = *(const float4*)(c + off);
        float co[4] = {cold.x, cold.y, cold.z, cold.w};
        float4 cnew;
        float* cn = (float*)&cnew;
        PackBF4 hp;
        #pragma unroll
        for (int e = 0; e < 4; ++e) {
            float f  = sigmoidf_fast(zv[0][e]);
            float ii = sigmoidf_fast(zv[1][e]);
            float gg = tanhf(zv[2][e]);
            float oo = sigmoidf_fast(zv[3][e]);
            float cv = gg * ii + co[e] * f;
            cn[e] = cv;
            hp.h[e] = f2bf(tanhf(cv) * oo);
        }
        *(float4*)(c + off) = cnew;
        *(unsigned long long*)(hout + off) = hp.u;
    }
}

// ---------------------------------------------------------------------------
// Final projection: logits(256x128) = h @ Wph^T + bph, via bf16 WMMA.
// Block = 128 threads (4 waves), wave w -> N subtile w. Grid = 16 x 2 = 32.
// ---------------------------------------------------------------------------
__global__ __launch_bounds__(128)
void proj_kernel(const u16* __restrict__ h, const u16* __restrict__ Wpb,
                 const float* __restrict__ bph, float* __restrict__ logits) {
    const int lane = threadIdx.x & 31;
    const int w    = threadIdx.x >> 5;
    const int bm   = blockIdx.x & 15;
    const int bn   = blockIdx.x >> 4;
    const int m_base = bm * 16;
    const int n_base = bn * 64 + w * 16;

    v8f acc = {};
    const u16* aptr = h   + (size_t)(m_base + (lane & 15)) * 1024 + ((lane >> 4) << 4);
    const u16* bptr = Wpb + (size_t)(n_base + (lane & 15)) * 1024 + ((lane >> 4) << 4);

    #pragma unroll 4
    for (int k0 = 0; k0 < 1024; k0 += 32) {
        FragBF a, b;
        a.q[0] = *(const uint4v*)(aptr + k0);
        a.q[1] = *(const uint4v*)(aptr + k0 + 8);
        b.q[0] = *(const uint4v*)(bptr + k0);
        b.q[1] = *(const uint4v*)(bptr + k0 + 8);
        acc = __builtin_amdgcn_wmma_f32_16x16x32_bf16(false, a.v, false, b.v,
                                                      (short)0, acc, false, false);
    }

    const int mh  = (lane >> 4) << 3;
    const int col = n_base + (lane & 15);
    #pragma unroll
    for (int v = 0; v < 8; ++v)
        logits[(size_t)(m_base + v + mh) * 128 + col] = acc[v] + bph[col];
}

// ---------------------------------------------------------------------------
// Softmax over 128 logits per batch row. Grid = 256 blocks x 128 threads.
// ---------------------------------------------------------------------------
__global__ __launch_bounds__(128)
void softmax_kernel(const float* __restrict__ logits, float* __restrict__ out) {
    __shared__ float red[128];
    const int b = blockIdx.x;
    const int col = threadIdx.x;
    float v = logits[(size_t)b * 128 + col];
    red[col] = v;
    __syncthreads();
    for (int s = 64; s > 0; s >>= 1) {
        if (col < s) red[col] = fmaxf(red[col], red[col + s]);
        __syncthreads();
    }
    float m = red[0];
    __syncthreads();
    float e = __expf(v - m);
    red[col] = e;
    __syncthreads();
    for (int s = 64; s > 0; s >>= 1) {
        if (col < s) red[col] += red[col + s];
        __syncthreads();
    }
    out[(size_t)b * 128 + col] = e / red[0];
}

// ---------------------------------------------------------------------------
extern "C" void kernel_launch(void* const* d_in, const int* in_sizes, int n_in,
                              void* d_out, int out_size, void* d_ws, size_t ws_size,
                              hipStream_t stream) {
    const float* x   = (const float*)d_in[0];
    const float* Wfx = (const float*)d_in[1];  const float* bfx = (const float*)d_in[2];
    const float* Wfh = (const float*)d_in[3];  const float* bfh = (const float*)d_in[4];
    const float* Wix = (const float*)d_in[5];  const float* bix = (const float*)d_in[6];
    const float* Wih = (const float*)d_in[7];  const float* bih = (const float*)d_in[8];
    const float* Wgx = (const float*)d_in[9];  const float* bgx = (const float*)d_in[10];
    const float* Wgh = (const float*)d_in[11]; const float* bgh = (const float*)d_in[12];
    const float* Wox = (const float*)d_in[13]; const float* box_ = (const float*)d_in[14];
    const float* Woh = (const float*)d_in[15]; const float* boh = (const float*)d_in[16];
    const float* Wph = (const float*)d_in[17]; const float* bph = (const float*)d_in[18];

    char* ws = (char*)d_ws;
    u16*   Whb    = (u16*)ws;                                          // 8 MB
    u16*   Wpb    = (u16*)(ws + (size_t)8 * 1024 * 1024);              // 256 KB
    u16*   hA     = (u16*)(ws + (size_t)8 * 1024 * 1024 + 256 * 1024); // 512 KB
    u16*   hB     = hA + (size_t)256 * 1024;                           // 512 KB
    float* c      = (float*)((char*)(hB + (size_t)256 * 1024));        // 1 MB
    float* logits = (float*)((char*)c + (size_t)256 * 1024 * 4);       // 128 KB
    float* Wxp    = (float*)((char*)logits + (size_t)256 * 128 * 4);   // 16 KB
    float* bsum   = (float*)((char*)Wxp + (size_t)4 * 1024 * 4);       // 16 KB

    prep_kernel<<<2048, 256, 0, stream>>>(Wfh, Wih, Wgh, Woh, Wph,
                                          Wfx, Wix, Wgx, Wox,
                                          bfx, bix, bgx, box_,
                                          bfh, bih, bgh, boh,
                                          Whb, Wpb, Wxp, bsum, hA, hB, c);

    for (int t = 0; t < 128; ++t) {
        const u16* hin = (t & 1) ? hB : hA;
        u16*      hout = (t & 1) ? hA : hB;
        lstm_step<<<512, 128, 0, stream>>>(t, x, hin, hout, c, Whb, Wxp, bsum);
    }
    // 128 steps: last step (t=127, odd) wrote hA.
    proj_kernel<<<32, 128, 0, stream>>>(hA, Wpb, bph, logits);
    softmax_kernel<<<256, 128, 0, stream>>>(logits, (float*)d_out);
}
